// ProtoSupConLoss_40226663694798
// MI455X (gfx1250) — matrix operand order repaired and checked by reading, not compiled, not run on hardware
//
#include <hip/hip_runtime.h>
#include <math.h>

// Problem constants (reference: B=8, S=2048, D=128, K=64)
#define Bn 8
#define Sn 2048
#define Dn 128
#define Kn 64
#define BS (Bn * Sn)
#define ROWS 128      // output rows per block
#define CHUNK 64      // B-columns staged per LDS chunk
#define JSPLIT 2      // column-range split for 2x grid parallelism
#define JLEN (Sn / JSPLIT)
#define LP 136        // padded LDS row: 128 + 8 bf16 -> 272B stride (16B aligned, conflict-free)

typedef __attribute__((ext_vector_type(16))) __bf16        v16bf;
typedef __attribute__((ext_vector_type(8)))  float         v8f;
typedef __attribute__((ext_vector_type(4)))  unsigned int  u32x4;
typedef __attribute__((ext_vector_type(8)))  int           i32x8;
typedef __attribute__((ext_vector_type(4)))  int           i32x4;

#if __has_builtin(__builtin_amdgcn_tensor_load_to_lds)
#define HAVE_TDM 1
#endif

__device__ __forceinline__ unsigned short f2bf(float x) {
  unsigned u = __float_as_uint(x);
  unsigned r = u + 0x7FFFu + ((u >> 16) & 1u);   // round-to-nearest-even
  return (unsigned short)(r >> 16);
}
__device__ __forceinline__ float bf2f(unsigned short h) {
  return __uint_as_float(((unsigned)h) << 16);
}

union V16U { v16bf v; u32x4 q[2]; };

// one WMMA operand = two contiguous 8x-bf16 runs in LDS (ds_load_b128 each)
__device__ __forceinline__ v16bf ldsv16(const unsigned short* p0, const unsigned short* p1) {
  V16U u;
  u.q[0] = *reinterpret_cast<const u32x4*>(p0);
  u.q[1] = *reinterpret_cast<const u32x4*>(p1);
  return u.v;
}

// CDNA5 async DMA (ASYNCcnt): per-lane global -> LDS copies, no VGPR data path
__device__ __forceinline__ void async_cp16(const void* g, void* l) {
  unsigned           loff = (unsigned)(size_t)l;
  unsigned long long ga   = (unsigned long long)(size_t)g;
  asm volatile("global_load_async_to_lds_b128 %0, %1, off" :: "v"(loff), "v"(ga) : "memory");
}
__device__ __forceinline__ void async_cp4(const void* g, void* l) {
  unsigned           loff = (unsigned)(size_t)l;
  unsigned long long ga   = (unsigned long long)(size_t)g;
  asm volatile("global_load_async_to_lds_b32 %0, %1, off" :: "v"(loff), "v"(ga) : "memory");
}
__device__ __forceinline__ void wait_async0() {
  asm volatile("s_wait_asynccnt 0" ::: "memory");
}
__device__ __forceinline__ void wait_tensor0() {
#if __has_builtin(__builtin_amdgcn_s_wait_tensorcnt)
  __builtin_amdgcn_s_wait_tensorcnt(0);
#else
  asm volatile("s_wait_tensorcnt 0" ::: "memory");
#endif
}

#ifdef HAVE_TDM
// TDM: move one CHUNK x 128 bf16 tile (rows of 256B, 8B elements) global->LDS.
// pad_interval=5 (256B) + pad_amount=3 (4 DWORDs = 16B) reproduces the LP=136 padded rows.
__device__ __forceinline__ void tdm_load_tile(const void* gsrc, unsigned lds_off) {
  unsigned long long ga = (unsigned long long)(size_t)gsrc;
  u32x4 g0;
  g0.x = 1u;                                               // count=1 (valid user D#)
  g0.y = lds_off;                                          // lds_addr
  g0.z = (unsigned)(ga & 0xFFFFFFFFu);                     // global_addr[31:0]
  g0.w = (unsigned)((ga >> 32) & 0x1FFFFFFu) | (2u << 30); // global_addr[56:32] | type=2
  i32x8 g1;
  g1[0] = (3 << 16) | (1 << 20) | (5 << 22) | (3 << 25);   // data_size=8B, pad_en, ivl=256B, amt=16B
  g1[1] = (32 << 16);                                      // tensor_dim0 = 32 units (one 256B row)
  g1[2] = 0;                                               // tensor_dim0 hi / tensor_dim1 lo
  g1[3] = 16 | (32 << 16);                                 // tensor_dim1 = 1<<20 rows, tile_dim0 = 32
  g1[4] = CHUNK;                                           // tile_dim1 = 64 rows, tile_dim2 = 0
  g1[5] = 32;                                              // tensor_dim0_stride = 32 units (contig rows)
  g1[6] = 0;
  g1[7] = 0;
  i32x4 z4 = {0, 0, 0, 0};                                 // groups 2/3 unused (2D tensor)
  i32x8 z8 = {0, 0, 0, 0, 0, 0, 0, 0};
  __builtin_amdgcn_tensor_load_to_lds(g0, g1, z4, z4, z8, 0);   // 6-arg toolchain variant
}
#endif

// ---------------------------------------------------------------- zero init
__global__ __launch_bounds__(256) void k_zero(float* out, int* hist) {
  int t = threadIdx.x;
  if (t == 0) out[0] = 0.f;
  hist[t] = 0;              // Bn*Kn == 512
  hist[t + 256] = 0;
}

// -------------------------------- split features into bf16 hi/lo (one pass)
__global__ __launch_bounds__(256) void k_convert(const float* __restrict__ f,
                                                 unsigned short* __restrict__ fhi,
                                                 unsigned short* __restrict__ flo) {
  int i = blockIdx.x * 256 + threadIdx.x;          // one float4 per thread
  float4 v = reinterpret_cast<const float4*>(f)[i];
  unsigned short h0 = f2bf(v.x), h1 = f2bf(v.y), h2 = f2bf(v.z), h3 = f2bf(v.w);
  unsigned short l0 = f2bf(v.x - bf2f(h0)), l1 = f2bf(v.y - bf2f(h1));
  unsigned short l2 = f2bf(v.z - bf2f(h2)), l3 = f2bf(v.w - bf2f(h3));
  uint2 hp, lp;
  hp.x = (unsigned)h0 | ((unsigned)h1 << 16);  hp.y = (unsigned)h2 | ((unsigned)h3 << 16);
  lp.x = (unsigned)l0 | ((unsigned)l1 << 16);  lp.y = (unsigned)l2 | ((unsigned)l3 << 16);
  reinterpret_cast<uint2*>(fhi)[i] = hp;
  reinterpret_cast<uint2*>(flo)[i] = lp;
}

// ----------------------------------------------- per-row |f|^2 + label hist
__global__ __launch_bounds__(256) void k_prep(const float* __restrict__ f,
                                              const int* __restrict__ lab,
                                              float* __restrict__ sq,
                                              int* __restrict__ hist) {
  int idx = blockIdx.x * 256 + threadIdx.x;
  const float4* fp = reinterpret_cast<const float4*>(f + (size_t)idx * Dn);
  float s = 0.f;
#pragma unroll
  for (int q = 0; q < Dn / 4; ++q) {
    float4 v = fp[q];
    s += v.x * v.x + v.y * v.y + v.z * v.z + v.w * v.w;
  }
  sq[idx] = s;
  int b = idx >> 11;        // idx / Sn
  atomicAdd(&hist[b * Kn + lab[idx]], 1);
}

// --------------------------------------------------------- fused gram kernel
// gram = hi*hi + hi*lo + lo*hi (split-fp32 via bf16 WMMA), fused with
// d2 = clamp((sqi+sqj-2*gram)/D, 0) and masked per-row accumulation.
// B tiles are double-buffered in LDS, filled by TDM (TENSORcnt) / async DMA
// (ASYNCcnt) overlapped with the WMMA work on the previous chunk.
__global__ __launch_bounds__(256) void k_gram(const unsigned short* __restrict__ fhi,
                                              const unsigned short* __restrict__ flo,
                                              const int* __restrict__ lab,
                                              const float* __restrict__ sq,
                                              float* __restrict__ negS,
                                              float* __restrict__ posS) {
  __shared__ alignas(16) unsigned short Ahi[ROWS][LP];
  __shared__ alignas(16) unsigned short Alo[ROWS][LP];
  __shared__ alignas(16) unsigned short Bhi[2][CHUNK][LP];
  __shared__ alignas(16) unsigned short Blo[2][CHUNK][LP];
  __shared__ float sqj[2][CHUNK];
  __shared__ int   jlabs[2][CHUNK];

  const int tid  = threadIdx.x;
  const int b    = blockIdx.y;
  const int row0 = blockIdx.x * ROWS;
  const int jbeg = blockIdx.z * JLEN;
  const int jend = jbeg + JLEN;
  const int base = b * Sn;
  const int wave = tid >> 5;

  // ---- stage A (this block's 128 rows, hi+lo) with plain b128 copies
  {
    const unsigned short* ah = fhi + (size_t)(base + row0) * Dn;
    const unsigned short* al = flo + (size_t)(base + row0) * Dn;
    for (int e = tid; e < ROWS * 16; e += 256) {         // 16 uint4 per 128-elem row
      int r = e >> 4, c = e & 15;
      *reinterpret_cast<u32x4*>(&Ahi[r][c * 8]) =
          reinterpret_cast<const u32x4*>(ah + (size_t)r * Dn)[c];
      *reinterpret_cast<u32x4*>(&Alo[r][c * 8]) =
          reinterpret_cast<const u32x4*>(al + (size_t)r * Dn)[c];
    }
  }

  // ---- chunk staging (TDM tile move if available, else per-lane async DMA)
  auto stage = [&](int buf, int c0) {
#ifdef HAVE_TDM
    if (wave == 0) {    // one wave drives the Tensor Data Mover (EXEC-independent)
      tdm_load_tile(fhi + (size_t)(base + c0) * Dn, (unsigned)(size_t)&Bhi[buf][0][0]);
      tdm_load_tile(flo + (size_t)(base + c0) * Dn, (unsigned)(size_t)&Blo[buf][0][0]);
    }
#else
    const unsigned short* gh = fhi + (size_t)(base + c0) * Dn;
    const unsigned short* gl = flo + (size_t)(base + c0) * Dn;
    for (int e = tid; e < CHUNK * 16; e += 256) {
      int r = e >> 4, c = e & 15;
      async_cp16(reinterpret_cast<const u32x4*>(gh + (size_t)r * Dn) + c, &Bhi[buf][r][c * 8]);
      async_cp16(reinterpret_cast<const u32x4*>(gl + (size_t)r * Dn) + c, &Blo[buf][r][c * 8]);
    }
#endif
    if (tid < CHUNK) {
      async_cp4(sq + base + c0 + tid,  &sqj[buf][tid]);
      async_cp4(lab + base + c0 + tid, &jlabs[buf][tid]);
    }
  };

  stage(0, jbeg);
  wait_async0();
  wait_tensor0();
  __syncthreads();

  const int lane = tid & 31;
  const int half = lane >> 4;   // 0: lanes 0-15, 1: lanes 16-31
  const int jn   = lane & 15;

  // fixed per-lane output-row metadata (C/D layout: VGPR r -> M=r+8*half, lane -> N)
  int gi[8]; float sqi[8]; int labi[8];
#pragma unroll
  for (int r = 0; r < 8; ++r) {
    gi[r]   = row0 + wave * 16 + r + half * 8;
    sqi[r]  = sq[base + gi[r]];
    labi[r] = lab[base + gi[r]];
  }

  // hoist A operands into registers for the whole j sweep
  // A layout: lane%16 -> M row; VGPR 0-3: K = 8*half + {0..7}; VGPR 4-7: K = +16
  const int arow = wave * 16 + jn;
  const int acb  = half * 8;
  v16bf aH[4], aL[4];
#pragma unroll
  for (int kc = 0; kc < 4; ++kc) {
    int k0 = kc * 32 + acb;
    aH[kc] = ldsv16(&Ahi[arow][k0], &Ahi[arow][k0 + 16]);
    aL[kc] = ldsv16(&Alo[arow][k0], &Alo[arow][k0 + 16]);
  }

  float negA[8], posA[8];
#pragma unroll
  for (int r = 0; r < 8; ++r) { negA[r] = 0.f; posA[r] = 0.f; }

  int cur = 0;
  for (int c0 = jbeg; c0 < jend; c0 += CHUNK) {
    const int nxt = cur ^ 1;

    // overlap: DMA the next chunk while computing this one
    if (c0 + CHUNK < jend) stage(nxt, c0 + CHUNK);

#pragma unroll
    for (int cb = 0; cb < CHUNK / 16; ++cb) {
      const int jt0  = cb * 16;
      // B layout: lane%16 -> N col; VGPR 0-3: K = 8*half + {0..7}; VGPR 4-7: K = +16
      const int brow = jt0 + jn;
      const int bcb  = half * 8;
      v8f acc = {};
#pragma unroll
      for (int kc = 0; kc < 4; ++kc) {
        int k0 = kc * 32 + bcb;
        v16bf bh = ldsv16(&Bhi[cur][brow][k0], &Bhi[cur][brow][k0 + 16]);
        v16bf bl = ldsv16(&Blo[cur][brow][k0], &Blo[cur][brow][k0 + 16]);
        acc = __builtin_amdgcn_wmma_f32_16x16x32_bf16(false, aH[kc], false, bh, (short)0, acc, false, false);
        acc = __builtin_amdgcn_wmma_f32_16x16x32_bf16(false, aH[kc], false, bl, (short)0, acc, false, false);
        acc = __builtin_amdgcn_wmma_f32_16x16x32_bf16(false, aL[kc], false, bh, (short)0, acc, false, false);
      }
      const int   jloc = jt0 + jn;
      const int   gj   = c0 + jloc;
      const float sqjv = sqj[cur][jloc];
      const int   labj = jlabs[cur][jloc];
#pragma unroll
      for (int r = 0; r < 8; ++r) {
        float d2 = fmaxf((sqi[r] + sqjv - 2.f * acc[r]) * (1.f / (float)Dn), 0.f);
        bool same = (labj == labi[r]);
        negA[r] += same ? 0.f : d2;                          // branchless v_cndmask + v_add
        posA[r] += (same && (gi[r] != gj)) ? d2 : 0.f;
      }
    }

    wait_async0();      // next chunk's per-lane DMA landed (this wave)
    wait_tensor0();     // next chunk's TDM tiles landed (issuing wave; no-op elsewhere)
    __syncthreads();    // all waves done with 'cur'; 'nxt' visible to all
    cur = nxt;
  }

  // reduce across the 16 lanes that share each output row
#pragma unroll
  for (int r = 0; r < 8; ++r) {
#pragma unroll
    for (int m = 1; m < 16; m <<= 1) {
      negA[r] += __shfl_xor(negA[r], m, 32);
      posA[r] += __shfl_xor(posA[r], m, 32);
    }
  }
  if (jn == 0) {
    const int slab = blockIdx.z * BS;
#pragma unroll
    for (int r = 0; r < 8; ++r) {
      negS[slab + base + gi[r]] = negA[r];
      posS[slab + base + gi[r]] = posA[r];
    }
  }
}

// ---------------------------------------- centroid terms + final reduction
__global__ __launch_bounds__(256) void k_finish(const float* __restrict__ f,
                                                const float* __restrict__ cen,
                                                const float* __restrict__ phi,
                                                const int* __restrict__ lab,
                                                const float* __restrict__ sq,
                                                const float* __restrict__ negS,
                                                const float* __restrict__ posS,
                                                const int* __restrict__ hist,
                                                float* __restrict__ out) {
  __shared__ float cent[Kn * Dn];   // 32 KB
  __shared__ float csq[Kn];
  __shared__ float phis[Kn];
  __shared__ float red[256];
  const int tid = threadIdx.x;

  for (int e = tid; e < Kn * Dn; e += 256) cent[e] = cen[e];
  __syncthreads();
  if (tid < Kn) {
    float s = 0.f;
#pragma unroll 16
    for (int d = 0; d < Dn; ++d) { float v = cent[tid * Dn + d]; s += v * v; }
    csq[tid]  = s;
    phis[tid] = phi[tid];
  }
  __syncthreads();

  const int idx = blockIdx.x * 256 + tid;
  const int b   = idx >> 11;
  const int myl = lab[idx];

  float4 fr[Dn / 4];
  const float4* fp = reinterpret_cast<const float4*>(f + (size_t)idx * Dn);
#pragma unroll
  for (int q = 0; q < Dn / 4; ++q) fr[q] = fp[q];
  const float sqv = sq[idx];

  float sumc = 0.f, pcent = 0.f;
  for (int k = 0; k < Kn; ++k) {
    const float4* cp = reinterpret_cast<const float4*>(&cent[k * Dn]);
    float dot = 0.f;
#pragma unroll
    for (int q = 0; q < Dn / 4; ++q) {
      float4 cv = cp[q];
      dot += fr[q].x * cv.x + fr[q].y * cv.y + fr[q].z * cv.z + fr[q].w * cv.w;
    }
    float cd2 = fmaxf((sqv + csq[k] - 2.f * dot) * (1.f / (float)Dn), 0.f) / phis[k];
    sumc += cd2;
    if (k == myl) pcent = cd2;
  }

  const float EULER = 2.718281828459045f;
  float negc = (sumc - pcent) * (1.f / (float)(Kn - 1));
  float ncl  = negc * logf(negc + EULER);

  int cnt    = hist[b * Kn + myl];
  int poscnt = cnt - 1;
  int negcnt = Sn - cnt;
  float possum = posS[idx] + posS[BS + idx];   // two j-split slabs
  float negsum = negS[idx] + negS[BS + idx];
  float posx = possum / (float)(poscnt > 1 ? poscnt : 1);
  float negx = negsum / (float)(negcnt > 1 ? negcnt : 1);
  float nsup = negx * logf(negx + EULER);

  // SIG = 0.5
  float pos_spcl = 0.5f * posx + 0.5f * pcent;
  float neg_spcl = 0.5f * nsup + 0.5f * ncl;
  float per      = pos_spcl - neg_spcl + 10.0f;   // EPSILON = 10
  float val      = (negcnt > 0) ? per : 0.f;

  red[tid] = val;
  __syncthreads();
  for (int s2 = 128; s2 > 0; s2 >>= 1) {
    if (tid < s2) red[tid] += red[tid + s2];
    __syncthreads();
  }
  if (tid == 0) atomicAdd(out, red[0] * (1.f / (float)Sn));
}

extern "C" void kernel_launch(void* const* d_in, const int* in_sizes, int n_in,
                              void* d_out, int out_size, void* d_ws, size_t ws_size,
                              hipStream_t stream) {
  const float* features  = (const float*)d_in[0];
  const float* centroids = (const float*)d_in[1];
  const float* phi       = (const float*)d_in[2];
  const int*   labels    = (const int*)d_in[3];
  float*       out       = (float*)d_out;

  // workspace: sq[BS] | negSum[2*BS] | posSum[2*BS] | hist[512] | fhi[BS*Dn] bf16 | flo[BS*Dn] bf16
  float*          sq   = (float*)d_ws;
  float*          negS = sq + BS;
  float*          posS = negS + JSPLIT * BS;
  int*            hist = (int*)(posS + JSPLIT * BS);
  unsigned short* fhi  = (unsigned short*)(hist + Bn * Kn);
  unsigned short* flo  = fhi + (size_t)BS * Dn;

  k_zero<<<1, 256, 0, stream>>>(out, hist);
  k_convert<<<(BS * (Dn / 4)) / 256, 256, 0, stream>>>(features, fhi, flo);
  k_prep<<<BS / 256, 256, 0, stream>>>(features, labels, sq, hist);
  k_gram<<<dim3(Sn / ROWS, Bn, JSPLIT), 256, 0, stream>>>(fhi, flo, labels, sq, negS, posS);
  k_finish<<<BS / 256, 256, 0, stream>>>(features, centroids, phi, labels, sq, negS, posS, hist, out);
}
